// SparseCodingLayer_AfterConv_56882546868255
// MI455X (gfx1250) — matrix operand order repaired and checked by reading, not compiled, not run on hardware
//
#include <hip/hip_runtime.h>

// ---------------------------------------------------------------------------
// Problem dims (from reference): x=[32,64,56,56]
//   POS = 32*56*56 = 100352 spatial positions
//   enc:  implicit im2col GEMM  [POS,576] x [576,256]  (channel-last input)
//   dec:  GEMM [POS,256] x [256,576]
//   red:  GEMM [POS,256] x [256,128]
// ---------------------------------------------------------------------------
enum : int {
  NB   = 32,
  CIN  = 64,
  HH   = 56,
  HW   = 56 * 56,            // 3136
  PH   = 58,                 // padded H/W
  PHW  = 58 * 58,            // 3364
  POS  = NB * HW,            // 100352
  NENC = 256,
  KENC = 576,                // 64*9
  NDEC = 576,
  NRED = 128,
  KACT = 256,
  MT   = 128,                // M tile per block
  MTILES = POS / MT,         // 784
  XPELEMS = NB * PHW * CIN,  // padded channel-last input elems = 6889472
  ASTRIDE = 136,             // LDS row stride for act tile (bank-spread, 16B mult)
};

typedef __attribute__((ext_vector_type(16))) __bf16 v16bf;
typedef __attribute__((ext_vector_type(8)))  __bf16 v8bf;
typedef __attribute__((ext_vector_type(8)))  float  v8f;

// ---------------- bf16 helpers (round-to-nearest-even) ---------------------
static __device__ __forceinline__ float bf2f(__bf16 h) {
  unsigned short s = __builtin_bit_cast(unsigned short, h);
  unsigned u = ((unsigned)s) << 16;
  return __builtin_bit_cast(float, u);
}
static __device__ __forceinline__ __bf16 f2bf(float f) {
  unsigned u = __builtin_bit_cast(unsigned, f);
  u += 0x7fffu + ((u >> 16) & 1u);
  unsigned short s = (unsigned short)(u >> 16);
  return __builtin_bit_cast(__bf16, s);
}
static __device__ __forceinline__ v16bf zfrag16() {
  v16bf v;
#pragma unroll
  for (int i = 0; i < 16; ++i) v[i] = f2bf(0.0f);
  return v;
}
// Build a 16-elem bf16 fragment from two contiguous 16-byte chunks.
static __device__ __forceinline__ v16bf ldfrag16(const __bf16* lo, const __bf16* hi) {
  v8bf a = *(const v8bf*)lo;
  v8bf b = *(const v8bf*)hi;
  return __builtin_shufflevector(a, b, 0, 1, 2, 3, 4, 5, 6, 7, 8, 9, 10, 11, 12, 13, 14, 15);
}
static __device__ __forceinline__ v8f wmma_bf16(v16bf a, v16bf b, v8f c) {
  return __builtin_amdgcn_wmma_f32_16x16x32_bf16(
      /*neg_a=*/false, a, /*neg_b=*/false, b,
      /*c_mod=*/(short)0, c, /*reuse_a=*/false, /*reuse_b=*/false);
}

// ---------------------------------------------------------------------------
// K1: per-channel BN stats of x (64 channels) -> scale/shift pairs
// ---------------------------------------------------------------------------
__global__ __launch_bounds__(256) void bnstats_kernel(
    const float* __restrict__ x, const float* __restrict__ g,
    const float* __restrict__ bb, float* __restrict__ st) {
  const int c = blockIdx.x, tid = threadIdx.x;
  float s = 0.f, q = 0.f;
  for (int b = 0; b < NB; ++b) {
    const float* p = x + (size_t)(b * CIN + c) * HW;
    for (int i = tid; i < HW; i += 256) {
      float v = p[i];
      s += v; q += v * v;
    }
  }
  __shared__ float rs[256], rq[256];
  rs[tid] = s; rq[tid] = q;
  __syncthreads();
  for (int off = 128; off > 0; off >>= 1) {
    if (tid < off) { rs[tid] += rs[tid + off]; rq[tid] += rq[tid + off]; }
    __syncthreads();
  }
  if (tid == 0) {
    const float n = (float)NB * (float)HW;
    float mu = rs[0] / n;
    float var = rq[0] / n - mu * mu;
    float sc = g[c] * rsqrtf(var + 1e-5f);
    st[c * 2 + 0] = sc;
    st[c * 2 + 1] = bb[c] - mu * sc;
  }
}

// K1b: zero the padded bf16 input buffer (writes 2 bf16 per thread as u32)
__global__ __launch_bounds__(256) void zero_kernel(unsigned* __restrict__ p) {
  p[blockIdx.x * 256 + threadIdx.x] = 0u;
}

// ---------------------------------------------------------------------------
// K2: xr = relu(bn(x)) as fp32 (NCHW, for aux target) and bf16 into the
//     zero-padded channel-last layout xcl[b][h+1][w+1][ci].
// ---------------------------------------------------------------------------
__global__ __launch_bounds__(256) void bnapply_kernel(
    const float* __restrict__ x, const float* __restrict__ st,
    float* __restrict__ xr, __bf16* __restrict__ xcl) {
  int idx = blockIdx.x * 256 + threadIdx.x;   // grid sized exactly
  int c = (idx / HW) & (CIN - 1);
  float v = fmaf(x[idx], st[c * 2 + 0], st[c * 2 + 1]);
  v = fmaxf(v, 0.0f);
  xr[idx] = v;
  int b = idx / (CIN * HW);
  int hw = idx % HW;
  int h = hw / HH, w = hw - h * HH;
  xcl[((size_t)(b * PH + (h + 1)) * PH + (w + 1)) * CIN + c] = f2bf(v);
}

// ---------------------------------------------------------------------------
// K3: fp32 -> bf16 weight conversion (layout preserved: [co][k], K-contig)
// ---------------------------------------------------------------------------
__global__ __launch_bounds__(256) void convert_kernel(
    const float* __restrict__ s, __bf16* __restrict__ d, int n) {
  int i = blockIdx.x * 256 + threadIdx.x;
  if (i < n) d[i] = f2bf(s[i]);
}

// K3b: enc weights: permute K order (ci,kh,kw) -> k' = (kh*3+kw)*64 + ci
__global__ __launch_bounds__(256) void convert_enc_kernel(
    const float* __restrict__ s, __bf16* __restrict__ d) {
  int i = blockIdx.x * 256 + threadIdx.x;   // grid covers NENC*KENC exactly
  int co = i / KENC, k = i - co * KENC;
  int ci = k / 9, r = k - ci * 9;
  d[(size_t)co * KENC + r * CIN + ci] = f2bf(s[i]);
}

// ---------------------------------------------------------------------------
// K4: enc implicit GEMM (channel-last): yenc[pos][256] = patches * wperm + b
//     block = 128 pos x 256 ch, 8 waves, wave = 128M x 32N (tall).
//     A fragments are direct global b128 loads (k'-run = contiguous ci),
//     software-pipelined one stage ahead of their WMMAs. Fully unrolled.
// ---------------------------------------------------------------------------
__global__ __launch_bounds__(256) void enc_gemm_kernel(
    const __bf16* __restrict__ xcl, const __bf16* __restrict__ wperm,
    const float* __restrict__ bias, float* __restrict__ yenc) {
  const int tid = threadIdx.x, lane = tid & 31, wid = tid >> 5;
  const int lrow = lane & 15, lhalf = lane >> 4;
  const int m0 = blockIdx.x * MT;
  const int wn = wid * 32;           // wave's 32-column slice of N=256

  // per-(mf,lane) base address of the patch origin in xcl (element units)
  int abase[8];
#pragma unroll
  for (int mf = 0; mf < 8; ++mf) {
    int pos = m0 + mf * 16 + lrow;
    int b = pos / HW, hw = pos - b * HW;
    int h = hw / HH, w = hw - h * HH;
    abase[mf] = ((b * PH + h) * PH + w) * CIN;
  }

  v8f acc[8][2];
#pragma unroll
  for (int i = 0; i < 8; ++i)
#pragma unroll
    for (int j = 0; j < 2; ++j)
#pragma unroll
      for (int e = 0; e < 8; ++e) acc[i][j][e] = 0.0f;

#pragma unroll
  for (int ks = 0; ks < KENC / 32; ++ks) {
    const int blk = ks >> 1;                 // (kh,kw) tap index 0..8
    const int kh = blk / 3, kw = blk - kh * 3;
    const int koffc = (kh * PH + kw) * CIN + (ks & 1) * 32 + lhalf * 8;

    // 2 B fragments for this wave's 32 columns (unique across the block)
    const __bf16* bp0 = wperm + (size_t)(wn + lrow) * KENC + ks * 32 + lhalf * 8;
    v16bf b0 = ldfrag16(bp0, bp0 + 16);
    const __bf16* bp1 = bp0 + (size_t)16 * KENC;
    v16bf b1 = ldfrag16(bp1, bp1 + 16);

    // A fragments pipelined one stage ahead of their WMMAs
    const __bf16* ap0 = xcl + abase[0] + koffc;
    v16bf a_cur = ldfrag16(ap0, ap0 + 16);
#pragma unroll
    for (int mf = 0; mf < 8; ++mf) {
      v16bf a_next = a_cur;
      if (mf < 7) {
        const __bf16* apn = xcl + abase[mf + 1] + koffc;
        a_next = ldfrag16(apn, apn + 16);
      }
      acc[mf][0] = wmma_bf16(a_cur, b0, acc[mf][0]);
      acc[mf][1] = wmma_bf16(a_cur, b1, acc[mf][1]);
      a_cur = a_next;
    }
  }

#pragma unroll
  for (int mf = 0; mf < 8; ++mf)
#pragma unroll
    for (int nf = 0; nf < 2; ++nf) {
      const int n = wn + nf * 16 + lrow;
      const float bv = bias[n];
      const int rb = m0 + mf * 16 + lhalf * 8;
#pragma unroll
      for (int r = 0; r < 8; ++r)
        yenc[(size_t)(rb + r) * NENC + n] = acc[mf][nf][r] + bv;
    }
}

// ---------------------------------------------------------------------------
// K5: channel top-K (K=32 of 256) per position; wave per row, radix-select
//     on |x| bit pattern. Writes masked activations as bf16 [pos][256].
// ---------------------------------------------------------------------------
__global__ __launch_bounds__(256) void topk_kernel(
    const float* __restrict__ yenc, __bf16* __restrict__ act) {
  const int lane = threadIdx.x & 31, wid = threadIdx.x >> 5;
  const int row = blockIdx.x * 8 + wid;
  const float* p = yenc + (size_t)row * NENC + lane * 8;
  float v[8];
  unsigned u[8];
#pragma unroll
  for (int j = 0; j < 8; ++j) {
    v[j] = p[j];
    u[j] = __builtin_bit_cast(unsigned, v[j]) & 0x7fffffffu;
  }
  // max thr with count(u >= thr) >= 32  == 32nd largest |value|
  unsigned thr = 0u;
  for (int bit = 30; bit >= 0; --bit) {
    unsigned cand = thr | (1u << bit);
    int cnt = 0;
#pragma unroll
    for (int j = 0; j < 8; ++j) cnt += (u[j] >= cand) ? 1 : 0;
#pragma unroll
    for (int off = 16; off > 0; off >>= 1) cnt += __shfl_xor(cnt, off, 32);
    if (cnt >= 32) thr = cand;
  }
  __bf16* q = act + (size_t)row * NENC + lane * 8;
#pragma unroll
  for (int j = 0; j < 8; ++j) q[j] = f2bf(u[j] >= thr ? v[j] : 0.0f);
}

// ---------------------------------------------------------------------------
// Dense GEMM core over K=256 bf16 activations. Wave = 128M x 16N (tall),
// A tile staged in LDS in two K-halves (128x128 each, conflict-free stride).
// Per k-step per wave: 1 global B frag + 8 LDS-fed WMMAs (A pipelined).
// ---------------------------------------------------------------------------
static __device__ __forceinline__ void act_gemm_core(
    const __bf16* __restrict__ act, const __bf16* __restrict__ w,
    __bf16* As, int Ncols, int m0, int nb, v8f acc[8]) {
  const int tid = threadIdx.x;
  const int lane = tid & 31, wid = tid >> 5;
  const int lrow = lane & 15, lhalf = lane >> 4;
#pragma unroll
  for (int i = 0; i < 8; ++i)
#pragma unroll
    for (int e = 0; e < 8; ++e) acc[i][e] = 0.0f;

  const int nglob = nb + wid * 16 + lrow;   // this wave's output column
  for (int half = 0; half < 2; ++half) {
    __syncthreads();
    // stage 128 rows x 128 K (coalesced b128 loads)
#pragma unroll
    for (int i = 0; i < 8; ++i) {
      int idx = i * 256 + tid;
      int row = idx >> 4, c16 = idx & 15;
      *(v8bf*)&As[row * ASTRIDE + c16 * 8] =
          *(const v8bf*)(act + (size_t)(m0 + row) * KACT + half * 128 + c16 * 8);
    }
    __syncthreads();
#pragma unroll
    for (int kk = 0; kk < 128; kk += 32) {
      v16bf bfr;
      if (nglob < Ncols) {
        const __bf16* bp = w + (size_t)nglob * KACT + half * 128 + kk + lhalf * 8;
        bfr = ldfrag16(bp, bp + 16);
      } else {
        bfr = zfrag16();
      }
      const __bf16* ab0 = &As[lrow * ASTRIDE + kk + lhalf * 8];
      v16bf a_cur = ldfrag16(ab0, ab0 + 16);
#pragma unroll
      for (int mf = 0; mf < 8; ++mf) {
        v16bf a_next = a_cur;
        if (mf < 7) {
          const __bf16* abn = &As[((mf + 1) * 16 + lrow) * ASTRIDE + kk + lhalf * 8];
          a_next = ldfrag16(abn, abn + 16);
        }
        acc[mf] = wmma_bf16(a_cur, bfr, acc[mf]);
        a_cur = a_next;
      }
    }
  }
}

// K6: dec GEMM -> decout bf16 [pos][576] (bias included, pre-BN)
__global__ __launch_bounds__(256) void dec_gemm_kernel(
    const __bf16* __restrict__ act, const __bf16* __restrict__ w,
    const float* __restrict__ bias, __bf16* __restrict__ decout) {
  __shared__ __bf16 As[MT * ASTRIDE];
  const int lane = threadIdx.x & 31, wid = threadIdx.x >> 5;
  const int lrow = lane & 15, lhalf = lane >> 4;
  const int m0 = blockIdx.x * MT;
  const int nb = blockIdx.y * 128;
  v8f acc[8];
  act_gemm_core(act, w, As, NDEC, m0, nb, acc);
  const int n = nb + wid * 16 + lrow;
  if (n < NDEC) {
    const float bv = bias[n];
#pragma unroll
    for (int mf = 0; mf < 8; ++mf) {
      const int rb = m0 + mf * 16 + lhalf * 8;
#pragma unroll
      for (int r = 0; r < 8; ++r)
        decout[(size_t)(rb + r) * NDEC + n] = f2bf(acc[mf][r] + bv);
    }
  }
}

// K10: red GEMM -> y fp32 in NCHW directly into d_out
__global__ __launch_bounds__(256) void red_gemm_kernel(
    const __bf16* __restrict__ act, const __bf16* __restrict__ w,
    const float* __restrict__ bias, float* __restrict__ out) {
  __shared__ __bf16 As[MT * ASTRIDE];
  const int lane = threadIdx.x & 31, wid = threadIdx.x >> 5;
  const int lrow = lane & 15, lhalf = lane >> 4;
  const int m0 = blockIdx.x * MT;
  v8f acc[8];
  act_gemm_core(act, w, As, NRED, m0, 0, acc);
  const int n = wid * 16 + lrow;   // output channel 0..127
  const float bv = bias[n];
#pragma unroll
  for (int mf = 0; mf < 8; ++mf) {
    const int rb = m0 + mf * 16 + lhalf * 8;
#pragma unroll
    for (int r = 0; r < 8; ++r) {
      const int row = rb + r;                 // position
      const int b = row / HW, hw = row - b * HW;
      out[((size_t)(b * NRED + n)) * HW + hw] = acc[mf][r] + bv;
    }
  }
}

// ---------------------------------------------------------------------------
// K7: per-channel partial sums/sumsq of decout (block = 128 rows)
// ---------------------------------------------------------------------------
__global__ __launch_bounds__(256) void decstats_kernel(
    const __bf16* __restrict__ decout, float* __restrict__ part) {
  const int blk = blockIdx.x, tid = threadIdx.x;
  const size_t row0 = (size_t)blk * MT;
  for (int c = tid; c < NDEC; c += 256) {
    float s = 0.f, q = 0.f;
    for (int r = 0; r < MT; ++r) {
      float v = bf2f(decout[(row0 + r) * NDEC + c]);
      s += v; q += v * v;
    }
    part[((size_t)blk * NDEC + c) * 2 + 0] = s;
    part[((size_t)blk * NDEC + c) * 2 + 1] = q;
  }
}

// K8: finalize aux-BN scale/shift per channel (576)
__global__ __launch_bounds__(256) void absn_final_kernel(
    const float* __restrict__ part, const float* __restrict__ g,
    const float* __restrict__ bb, float* __restrict__ absn) {
  const int tid = threadIdx.x;
  for (int c = tid; c < NDEC; c += 256) {
    float s = 0.f, q = 0.f;
    for (int blk = 0; blk < MTILES; ++blk) {
      s += part[((size_t)blk * NDEC + c) * 2 + 0];
      q += part[((size_t)blk * NDEC + c) * 2 + 1];
    }
    const float n = (float)POS;
    float mu = s / n;
    float var = q / n - mu * mu;
    float sc = g[c] * rsqrtf(var + 1e-5f);
    absn[c * 2 + 0] = sc;
    absn[c * 2 + 1] = bb[c] - mu * sc;
  }
}

// ---------------------------------------------------------------------------
// K9: aux loss: relu(bn(decout)) vs im2col target from xr -> block partials
// ---------------------------------------------------------------------------
__global__ __launch_bounds__(256) void loss_kernel(
    const __bf16* __restrict__ decout, const float* __restrict__ absn,
    const float* __restrict__ xr, float* __restrict__ part) {
  const int blk = blockIdx.x, tid = threadIdx.x;
  const int row0 = blk * MT;
  float accv = 0.f;
  for (int idx = tid; idx < MT * NDEC; idx += 256) {
    int r = idx / NDEC, c = idx - r * NDEC;
    int pos = row0 + r;
    float a = bf2f(decout[(size_t)pos * NDEC + c]);
    a = fmaf(a, absn[c * 2 + 0], absn[c * 2 + 1]);
    a = fmaxf(a, 0.0f);
    int ci = c / 9, rr = c - ci * 9;
    int kh = rr / 3, kw = rr - kh * 3;
    int b = pos / HW, hw = pos - b * HW;
    int h = hw / HH, w = hw - h * HH;
    int ih = h + kh - 1, iw = w + kw - 1;
    float t = 0.f;
    if ((unsigned)ih < (unsigned)HH && (unsigned)iw < (unsigned)HH)
      t = xr[((size_t)(b * CIN + ci) * HH + ih) * HH + iw];
    float d = t - a;
    accv += d * d;
  }
  __shared__ float red[256];
  red[tid] = accv;
  __syncthreads();
  for (int off = 128; off > 0; off >>= 1) {
    if (tid < off) red[tid] += red[tid + off];
    __syncthreads();
  }
  if (tid == 0) part[blk] = red[0];
}

__global__ __launch_bounds__(256) void loss_final_kernel(
    const float* __restrict__ part, float* __restrict__ out) {
  const int tid = threadIdx.x;
  float s = 0.f;
  for (int i = tid; i < MTILES; i += 256) s += part[i];
  __shared__ float red[256];
  red[tid] = s;
  __syncthreads();
  for (int off = 128; off > 0; off >>= 1) {
    if (tid < off) red[tid] += red[tid + off];
    __syncthreads();
  }
  if (tid == 0) out[0] = red[0] * (1.0f / ((float)POS * (float)NDEC));
}

// ---------------------------------------------------------------------------
extern "C" void kernel_launch(void* const* d_in, const int* in_sizes, int n_in,
                              void* d_out, int out_size, void* d_ws, size_t ws_size,
                              hipStream_t stream) {
  (void)in_sizes; (void)n_in; (void)out_size; (void)ws_size;
  const float* x     = (const float*)d_in[0];
  const float* bn_g  = (const float*)d_in[1];
  const float* bn_b  = (const float*)d_in[2];
  const float* enc_w = (const float*)d_in[3];
  const float* enc_b = (const float*)d_in[4];
  const float* dec_w = (const float*)d_in[5];
  const float* dec_b = (const float*)d_in[6];
  const float* abn_g = (const float*)d_in[7];
  const float* abn_b = (const float*)d_in[8];
  const float* red_w = (const float*)d_in[9];
  const float* red_b = (const float*)d_in[10];
  float* out = (float*)d_out;

  size_t off = 0;
  auto wsp = [&](size_t bytes) {
    char* p = (char*)d_ws + off;
    off += (bytes + 255) & ~(size_t)255;
    return (void*)p;
  };
  float*  statsx  = (float*)wsp((size_t)CIN * 2 * 4);
  float*  xr      = (float*)wsp((size_t)NB * CIN * HW * 4);
  __bf16* xcl     = (__bf16*)wsp((size_t)XPELEMS * 2);         // padded ch-last
  __bf16* encperm = (__bf16*)wsp((size_t)NENC * KENC * 2);
  __bf16* decbf   = (__bf16*)wsp((size_t)NDEC * KACT * 2);
  __bf16* redbf   = (__bf16*)wsp((size_t)NRED * KACT * 2);
  float*  yenc    = (float*)wsp((size_t)POS * NENC * 4);
  __bf16* actb    = (__bf16*)wsp((size_t)POS * NENC * 2);
  __bf16* decout  = (__bf16*)wsp((size_t)POS * NDEC * 2);
  float*  decpart = (float*)wsp((size_t)MTILES * NDEC * 2 * 4);
  float*  absn    = (float*)wsp((size_t)NDEC * 2 * 4);
  float*  losspart= (float*)wsp((size_t)MTILES * 4);

  // 1. BN stats of x; zero padded channel-last input buffer
  bnstats_kernel<<<CIN, 256, 0, stream>>>(x, bn_g, bn_b, statsx);
  zero_kernel<<<(XPELEMS / 2) / 256, 256, 0, stream>>>((unsigned*)xcl);
  // 2. xr = relu(bn(x)), fp32 NCHW + padded channel-last bf16
  bnapply_kernel<<<(NB * CIN * HW) / 256, 256, 0, stream>>>(x, statsx, xr, xcl);
  // 3. weights -> bf16 (enc: permuted K order; dec/red: layout preserved)
  convert_enc_kernel<<<(NENC * KENC) / 256, 256, 0, stream>>>(enc_w, encperm);
  convert_kernel<<<(NDEC * KACT + 255) / 256, 256, 0, stream>>>(dec_w, decbf, NDEC * KACT);
  convert_kernel<<<(NRED * KACT + 255) / 256, 256, 0, stream>>>(red_w, redbf, NRED * KACT);
  // 4. enc implicit GEMM (channel-last) -> yenc [pos][256]
  enc_gemm_kernel<<<MTILES, 256, 0, stream>>>(xcl, encperm, enc_b, yenc);
  // 5. channel top-K mask -> act bf16 [pos][256]
  topk_kernel<<<POS / 8, 256, 0, stream>>>(yenc, actb);
  // 6. dec GEMM -> decout bf16 [pos][576]
  dec_gemm_kernel<<<dim3(MTILES, (NDEC + 127) / 128), 256, 0, stream>>>(actb, decbf, dec_b, decout);
  // 7-8. aux BN stats over decout
  decstats_kernel<<<MTILES, 256, 0, stream>>>(decout, decpart);
  absn_final_kernel<<<1, 256, 0, stream>>>(decpart, abn_g, abn_b, absn);
  // 9. aux loss
  loss_kernel<<<MTILES, 256, 0, stream>>>(decout, absn, xr, losspart);
  loss_final_kernel<<<1, 256, 0, stream>>>(losspart, out + (size_t)POS * NRED);
  // 10. red GEMM -> y NCHW fp32 in d_out
  red_gemm_kernel<<<MTILES, 256, 0, stream>>>(actb, redbf, red_b, out);
}
